// SimpleNoisyTopKRouter_33509334844053
// MI455X (gfx1250) — matrix operand order repaired and checked by reading, not compiled, not run on hardware
//
#include <hip/hip_runtime.h>

// ---------------------------------------------------------------------------
// MoE router for MI455X (gfx1250, wave32, WMMA):
//   logits[t,e] = sum_d x[t,d]*W[e,d] + b[e]
//   top-2 over e, normalized = softmax over the two selected logits
// GEMM is [16384 x 2048] x [2048 x 64] -> memory bound (134MB of x @ 23.3TB/s
// ~= 5.8us). bf16 WMMA (16x16x32, f32 accum) keeps matrix-core time << that.
// W (512KB) is pre-converted to bf16 once in d_ws so the hot loop's B
// fragments come straight from b128 loads (no per-iter VALU cvt -> fewer
// WMMA->VALU WAR hazard NOPs, 2/3 less loop VALU, half the W L2 traffic).
// ---------------------------------------------------------------------------

typedef __attribute__((ext_vector_type(16))) __bf16 v16bf;
typedef __attribute__((ext_vector_type(8)))  __bf16 v8bf;
typedef __attribute__((ext_vector_type(8)))  float  v8f;
typedef __attribute__((ext_vector_type(4)))  float  v4f;

#define D_MODEL          2048
#define NEXP             64
#define TOKENS_PER_WAVE  32
#define WAVES_PER_BLOCK  4
#define TOKENS_PER_BLOCK (TOKENS_PER_WAVE * WAVES_PER_BLOCK)
#define LDS_PITCH        65   // 64 + 1 pad -> conflict-free column reads
#define W_ELEMS          (NEXP * D_MODEL)

// ----------------------------- W pre-convert -------------------------------
// Row-major bf16 copy of W. 131072 elements, 8 per thread.
__global__ __launch_bounds__(256)
void convert_w_kernel(const float* __restrict__ W, __bf16* __restrict__ Wb) {
  const int i = (blockIdx.x * 256 + threadIdx.x) * 8;
  if (i + 8 <= W_ELEMS) {
    v4f a = *(const v4f*)(W + i);
    v4f b = *(const v4f*)(W + i + 4);
    v8bf o;
#pragma unroll
    for (int j = 0; j < 4; ++j) {
      o[j]     = (__bf16)a[j];
      o[4 + j] = (__bf16)b[j];
    }
    *(v8bf*)(Wb + i) = o;
  }
}

// ------------------------------ fragments ----------------------------------
// A fragment (this code's contiguous k order, same as B): lane half h = lane>>4.
// elements 0..15 <- K = h*16 + 0..15 (base[0..15])
// caller passes base = rowptr + k0 + h*16. x is streamed once -> NT loads.
__device__ __forceinline__ v16bf load_frag_a(const float* base) {
  v4f p0 = __builtin_nontemporal_load((const v4f*)(base + 0));
  v4f p1 = __builtin_nontemporal_load((const v4f*)(base + 4));
  v4f p2 = __builtin_nontemporal_load((const v4f*)(base + 8));
  v4f p3 = __builtin_nontemporal_load((const v4f*)(base + 12));
  v16bf f;
#pragma unroll
  for (int i = 0; i < 4; ++i) {
    f[i]      = (__bf16)p0[i];
    f[4 + i]  = (__bf16)p1[i];
    f[8 + i]  = (__bf16)p2[i];
    f[12 + i] = (__bf16)p3[i];
  }
  return f;
}

// B fragment (this code's contiguous k order, same as A): lane l holds column N = l&15,
// elements 0..15 <- K = (l>>4)*16 + 0..15, contiguous in memory since W is
// [expert][d] row-major. f32 fallback path (used only if ws too small).
__device__ __forceinline__ v16bf load_frag_b_f32(const float* base) {
  v4f p0 = *(const v4f*)(base + 0);
  v4f p1 = *(const v4f*)(base + 4);
  v4f p2 = *(const v4f*)(base + 8);
  v4f p3 = *(const v4f*)(base + 12);
  v16bf f;
#pragma unroll
  for (int i = 0; i < 4; ++i) {
    f[i]      = (__bf16)p0[i];
    f[4 + i]  = (__bf16)p1[i];
    f[8 + i]  = (__bf16)p2[i];
    f[12 + i] = (__bf16)p3[i];
  }
  return f;
}

// ------------------------------ main kernel --------------------------------
template <bool PRECONV>
__global__ __launch_bounds__(TOKENS_PER_BLOCK)
void router_topk_kernel(const float* __restrict__ x,
                        const float* __restrict__ Wf,     // f32 W (fallback)
                        const __bf16* __restrict__ Wb,    // bf16 W (fast path)
                        const float* __restrict__ bias,
                        float* __restrict__ out,
                        int ntok) {
  __shared__ float lds_tile[WAVES_PER_BLOCK * TOKENS_PER_WAVE * LDS_PITCH];
  __shared__ float lds_bias[NEXP];

  const int tid  = threadIdx.x;
  const int lane = tid & 31;
  const int wave = tid >> 5;

  if (tid < NEXP) lds_bias[tid] = bias[tid];
  __syncthreads();

  const int kh  = lane >> 4;   // lane half (0/1)
  const int r16 = lane & 15;

  const int tok0 = blockIdx.x * TOKENS_PER_BLOCK + wave * TOKENS_PER_WAVE;

  // A row base pointers for the two 16-row M tiles of this wave
  const float* arow0 = x + (size_t)(tok0 + r16) * D_MODEL + kh * 16;
  const float* arow1 = arow0 + (size_t)16 * D_MODEL;
  // B base pointers: this lane's expert column within each 16-expert N tile
  const __bf16* browb = Wb + (size_t)r16 * D_MODEL + kh * 16;
  const float*  browf = Wf + (size_t)r16 * D_MODEL + kh * 16;

  v8f acc[2][4];
#pragma unroll
  for (int mt = 0; mt < 2; ++mt)
#pragma unroll
    for (int nt = 0; nt < 4; ++nt)
      acc[mt][nt] = (v8f){};

#pragma unroll 2
  for (int k0 = 0; k0 < D_MODEL; k0 += 32) {
    v16bf a0 = load_frag_a(arow0 + k0);
    v16bf a1 = load_frag_a(arow1 + k0);
#pragma unroll
    for (int nt = 0; nt < 4; ++nt) {
      v16bf bt;
      if constexpr (PRECONV) {
        // 32B of contiguous bf16 -> two global_load_b128, no VALU cvt.
        // Address is 32B-aligned (r16*2048*2, kh*32, nt*65536, k0*2).
        bt = *(const v16bf*)(browb + (size_t)nt * 16 * D_MODEL + k0);
      } else {
        bt = load_frag_b_f32(browf + (size_t)nt * 16 * D_MODEL + k0);
      }
      acc[0][nt] = __builtin_amdgcn_wmma_f32_16x16x32_bf16(
          false, a0, false, bt, (short)0, acc[0][nt], false, false);
      acc[1][nt] = __builtin_amdgcn_wmma_f32_16x16x32_bf16(
          false, a1, false, bt, (short)0, acc[1][nt], false, false);
    }
  }

  // Spill 32x64 logit tile to LDS. C/D layout: VGPR r, lane l ->
  // M = r + 8*(l>=16), N = l&15 within a 16x16 tile.
  float* tile = &lds_tile[wave * TOKENS_PER_WAVE * LDS_PITCH];
#pragma unroll
  for (int mt = 0; mt < 2; ++mt)
#pragma unroll
    for (int nt = 0; nt < 4; ++nt)
#pragma unroll
      for (int r = 0; r < 8; ++r) {
        const int M = mt * 16 + kh * 8 + r;
        const int N = nt * 16 + r16;
        tile[M * LDS_PITCH + N] = acc[mt][nt][r];
      }
  __syncthreads();

  // Epilogue: each lane owns one token; scan 64 logits (+bias), keep top-2.
  // Bank-conflict-free: bank(lane,e) = (lane*65 + e) % 64 = (lane+e) % 64.
  const float* row = &tile[lane * LDS_PITCH];
  float best1 = -3.402823466e38f, best2 = -3.402823466e38f;
  int i1 = 0, i2 = 0;
  for (int e = 0; e < NEXP; ++e) {
    const float v = row[e] + lds_bias[e];
    if (v > best1) {            // strict '>': earliest index wins ties (jax top_k)
      best2 = best1; i2 = i1;
      best1 = v;     i1 = e;
    } else if (v > best2) {
      best2 = v;     i2 = e;
    }
  }

  // normalized = softmax over just the top-2 logits (global Z cancels)
  const float e2  = __expf(best2 - best1);   // <= 1, numerically safe
  const float inv = 1.0f / (1.0f + e2);

  const int t = tok0 + lane;
  if (t < ntok) {
    out[2 * t + 0] = inv;        // weight of top-1
    out[2 * t + 1] = e2 * inv;   // weight of top-2
    float* oidx = out + (size_t)2 * ntok;  // second tuple element, flat-concat
    oidx[2 * t + 0] = (float)i1;
    oidx[2 * t + 1] = (float)i2;
  }
}

extern "C" void kernel_launch(void* const* d_in, const int* in_sizes, int n_in,
                              void* d_out, int out_size, void* d_ws, size_t ws_size,
                              hipStream_t stream) {
  (void)out_size; (void)n_in;
  const float* x    = (const float*)d_in[0];
  const float* W    = (const float*)d_in[1];
  const float* bias = (const float*)d_in[2];
  float* out        = (float*)d_out;

  const int ntok = in_sizes[0] / D_MODEL;          // 4*4096 = 16384
  const int nblk = (ntok + TOKENS_PER_BLOCK - 1) / TOKENS_PER_BLOCK;

  const size_t wbytes = (size_t)W_ELEMS * sizeof(__bf16);  // 256 KB
  if (d_ws != nullptr && ws_size >= wbytes) {
    __bf16* Wb = (__bf16*)d_ws;
    convert_w_kernel<<<W_ELEMS / (256 * 8), 256, 0, stream>>>(W, Wb);
    router_topk_kernel<true><<<nblk, TOKENS_PER_BLOCK, 0, stream>>>(
        x, W, Wb, bias, out, ntok);
  } else {
    router_topk_kernel<false><<<nblk, TOKENS_PER_BLOCK, 0, stream>>>(
        x, W, (const __bf16*)nullptr, bias, out, ntok);
  }
}
